// SWINTransformerEncoder_49512382988372
// MI455X (gfx1250) — matrix operand order repaired and checked by reading, not compiled, not run on hardware
//
#include <hip/hip_runtime.h>
#include <math.h>

// ---------------- types / constants ----------------
typedef _Float16 half_t;
typedef __attribute__((ext_vector_type(8)))  _Float16 v8h;
typedef __attribute__((ext_vector_type(16))) _Float16 v16h;
typedef __attribute__((ext_vector_type(8)))  float    v8f;

#define MB      16      // batch
#define GH      64
#define GWD     64
#define GD      256
#define GNT     2048
#define GHEADS  8
#define GHD     32
#define GFF     512
#define GWS     8
#define GK      9
#define NLAYERS 2
#define GRID_TOK (GH*GWD)   // 4096 tokens per image

#define MODE_F16  0
#define MODE_GELU 1
#define MODE_RES  2

// ---------------- WMMA fragment helpers ----------------
// A (16x32 f16): lanes 0-15 row=l, K k0..k0+15;
// lanes 16-31 row=l-16, K k0+16..k0+31.
__device__ __forceinline__ v16h load_afrag(const half_t* base, int ld, int r0, int k0) {
  int l = threadIdx.x & 31;
  const half_t* p = base + (size_t)(r0 + (l & 15)) * ld + (k0 + ((l >> 4) << 4));
  return *(const v16h*)p;
}
// B (32x16 f16), loaded from row-major "B-transposed" storage BT[n][k]:
// lane picks column n = l&15; lanes 0-15 hold K k0..k0+15, lanes 16-31 K k0+16..k0+31.
__device__ __forceinline__ v16h load_bfrag(const half_t* base, int ld, int n0, int k0) {
  int l = threadIdx.x & 31;
  const half_t* p = base + (size_t)(n0 + (l & 15)) * ld + (k0 + ((l >> 4) << 4));
  return *(const v16h*)p;
}
__device__ __forceinline__ v8f wmma16(v16h a, v16h b, v8f c) {
  return __builtin_amdgcn_wmma_f32_16x16x32_f16(false, a, false, b, (short)0, c,
                                                false, false);
}

__device__ __forceinline__ float gelu_tanh(float x) {
  float x3 = x * x * x;
  return 0.5f * x * (1.f + tanhf(0.7978845608028654f * (x + 0.044715f * x3)));
}
__device__ __forceinline__ float wave_sum32(float v) {
  v += __shfl_xor(v, 16); v += __shfl_xor(v, 8); v += __shfl_xor(v, 4);
  v += __shfl_xor(v, 2);  v += __shfl_xor(v, 1);
  return v;
}

// LayerNorm of one 256-wide row by one wave; writes f16.
__device__ __forceinline__ void ln_row_f16(const float* src, const float* g,
                                           const float* b, half_t* dst, int lane) {
  float x[8]; float s = 0.f;
  #pragma unroll
  for (int i = 0; i < 8; i++) { x[i] = src[lane * 8 + i]; s += x[i]; }
  s = wave_sum32(s);
  float mu = s * (1.f / (float)GD);
  float var = 0.f;
  #pragma unroll
  for (int i = 0; i < 8; i++) { float d = x[i] - mu; var += d * d; }
  var = wave_sum32(var) * (1.f / (float)GD);
  float rstd = rsqrtf(var + 1e-5f);
  #pragma unroll
  for (int i = 0; i < 8; i++) {
    int c = lane * 8 + i;
    dst[c] = (half_t)((x[i] - mu) * rstd * g[c] + b[c]);
  }
}

// ---------------- small kernels ----------------
// fp32 W[din,dout] -> f16 WT[dout,din]
__global__ void wcast_t(const float* __restrict__ W, half_t* __restrict__ WT,
                        int din, int dout) {
  int e = blockIdx.x * 256 + threadIdx.x;
  if (e >= din * dout) return;
  int n = e / din, k = e - n * din;
  WT[(size_t)n * din + k] = (half_t)W[(size_t)k * dout + n];
}

// candidate 3x3 window == top-9 set (K == SIDE^2, softmax is order-invariant)
__global__ void idx_kernel(const float* __restrict__ xt, int* __restrict__ idxb) {
  int tt = blockIdx.x * 256 + threadIdx.x;           // flat (m, t)
  if (tt >= MB * GNT) return;
  const float spacing = 2.f / 63.f;
  float ni = floorf((xt[tt * 2 + 0] + 1.f) / spacing + 0.5f);
  float nj = floorf((xt[tt * 2 + 1] + 1.f) / spacing + 0.5f);
  int ii = (int)fminf(fmaxf(ni - 1.f, 0.f), 61.f);
  int jj = (int)fminf(fmaxf(nj - 1.f, 0.f), 61.f);
  #pragma unroll
  for (int oi = 0; oi < 3; oi++)
    #pragma unroll
    for (int oj = 0; oj < 3; oj++)
      idxb[(size_t)tt * GK + oi * 3 + oj] = (ii + oi) * GWD + (jj + oj);
}

// grid -> window-major copy (with roll by shift) + LN(f16). 8 waves/block, 1 row/wave.
__global__ void win_gather_ln(const float* __restrict__ zc_m, float* __restrict__ zw,
                              half_t* __restrict__ a16, const float* __restrict__ g,
                              const float* __restrict__ b, int shift) {
  int wave = threadIdx.x >> 5, lane = threadIdx.x & 31;
  int r = blockIdx.x * 8 + wave;                    // 0..4095 window-major
  int win = r >> 6, pos = r & 63;
  int wh = win >> 3, ww = win & 7, ph = pos >> 3, pw = pos & 7;
  int h = (wh * GWS + ph + shift) & 63;
  int w = (ww * GWS + pw + shift) & 63;
  const float* src = zc_m + ((size_t)h * GWD + w) * GD;
  float* dz = zw + (size_t)r * GD;
  #pragma unroll
  for (int i = 0; i < 8; i++) { int c = lane * 8 + i; dz[c] = src[c]; }
  ln_row_f16(src, g, b, a16 + (size_t)r * GD, lane);
}

__global__ void win_scatter(const float* __restrict__ zw, float* __restrict__ zc_m,
                            int shift) {
  int e = blockIdx.x * 256 + threadIdx.x;           // over 4096*256
  int r = e >> 8, c = e & 255;
  int win = r >> 6, pos = r & 63;
  int wh = win >> 3, ww = win & 7, ph = pos >> 3, pw = pos & 7;
  int h = (wh * GWS + ph + shift) & 63;
  int w = (ww * GWS + pw + shift) & 63;
  zc_m[((size_t)h * GWD + w) * GD + c] = zw[(size_t)r * GD + c];
}

__global__ void ln_rows(const float* __restrict__ x, const float* __restrict__ g,
                        const float* __restrict__ b, half_t* __restrict__ out) {
  int wave = threadIdx.x >> 5, lane = threadIdx.x & 31;
  int r = blockIdx.x * 8 + wave;
  ln_row_f16(x + (size_t)r * GD, g, b, out + (size_t)r * GD, lane);
}

__global__ void gather_ln_kv(const float* __restrict__ zc_flat, const int* __restrict__ idx,
                             const float* __restrict__ g, const float* __restrict__ b,
                             half_t* __restrict__ out) {
  int wave = threadIdx.x >> 5, lane = threadIdx.x & 31;
  int r = blockIdx.x * 8 + wave;                    // 0..nt*K-1
  ln_row_f16(zc_flat + (size_t)idx[r] * GD, g, b, out + (size_t)r * GD, lane);
}

// ---------------- WMMA GEMM: Y[rows,N] = A16[rows,KD] x W[KD,N] (+bias) ----------------
// One wave computes a 32x64 output tile: 2 A-frags x 4 B-frags -> 8 WMMAs per k-step.
// KD is a compile-time constant (256 or 512) so the k-loop fully unrolls and the
// loads pipeline ahead of the v_wmma chain. WT16 is W transposed [N][KD] f16.
template<int MODE, int KD>
__global__ __launch_bounds__(32) void gemm_wmma(const half_t* __restrict__ A,
                                                const half_t* __restrict__ WT,
                                                const float* __restrict__ bias,
                                                int Ncols,
                                                half_t* __restrict__ out16,
                                                float* __restrict__ out32) {
  int n0 = blockIdx.x * 64;
  int m0 = blockIdx.y * 32;
  v8f acc[2][4] = {};
  #pragma unroll
  for (int k0 = 0; k0 < KD; k0 += 32) {
    v16h a0 = load_afrag(A, KD, m0,      k0);
    v16h a1 = load_afrag(A, KD, m0 + 16, k0);
    #pragma unroll
    for (int nt = 0; nt < 4; nt++) {
      v16h b = load_bfrag(WT, KD, n0 + nt * 16, k0);
      acc[0][nt] = wmma16(a0, b, acc[0][nt]);
      acc[1][nt] = wmma16(a1, b, acc[1][nt]);
    }
  }
  int lane = threadIdx.x & 31;
  #pragma unroll
  for (int mt = 0; mt < 2; mt++)
    #pragma unroll
    for (int nt = 0; nt < 4; nt++) {
      int col = n0 + nt * 16 + (lane & 15);
      float bv = bias[col];
      #pragma unroll
      for (int j = 0; j < 8; j++) {
        int row = m0 + mt * 16 + j + ((lane >> 4) << 3);   // ISA C layout
        float v = acc[mt][nt][j] + bv;
        if (MODE == MODE_GELU) v = gelu_tanh(v);
        if (MODE == MODE_RES)  out32[(size_t)row * Ncols + col] += v;
        else                   out16[(size_t)row * Ncols + col] = (half_t)v;
      }
    }
}

// ---------------- SWIN window attention: 1 block/window, 1 wave/head ----------------
__global__ __launch_bounds__(256) void window_attn_kernel(const half_t* __restrict__ q16,
                                                          const half_t* __restrict__ k16,
                                                          const half_t* __restrict__ v16,
                                                          half_t* __restrict__ o16) {
  __shared__ half_t pbuf[GHEADS][16][64];           // softmax probs (A operand for PV)
  __shared__ half_t vT[GHEADS][GHD][64];            // V transposed per head
  int hd = threadIdx.x >> 5, lane = threadIdx.x & 31;
  int base = blockIdx.x * 64;
  const float scale = 0.17677669529663687f;         // 1/sqrt(32)

  // head-local V transpose into LDS (same wave produces & consumes)
  for (int key = 0; key < 64; key++)
    vT[hd][lane][key] = v16[(size_t)(base + key) * GD + hd * GHD + lane];

  for (int qr = 0; qr < 4; qr++) {
    v16h qa = load_afrag(q16, GD, base + qr * 16, hd * GHD);
    v8f s0 = {}, s1 = {}, s2 = {}, s3 = {};
    s0 = wmma16(qa, load_bfrag(k16, GD, base +  0, hd * GHD), s0);
    s1 = wmma16(qa, load_bfrag(k16, GD, base + 16, hd * GHD), s1);
    s2 = wmma16(qa, load_bfrag(k16, GD, base + 32, hd * GHD), s2);
    s3 = wmma16(qa, load_bfrag(k16, GD, base + 48, hd * GHD), s3);

    // row softmax in C layout: row j(+8) lives in lanes of one 16-group
    #pragma unroll
    for (int j = 0; j < 8; j++) {
      float a0 = s0[j], a1 = s1[j], a2 = s2[j], a3 = s3[j];
      float mx = fmaxf(fmaxf(a0, a1), fmaxf(a2, a3));
      mx = fmaxf(mx, __shfl_xor(mx, 8)); mx = fmaxf(mx, __shfl_xor(mx, 4));
      mx = fmaxf(mx, __shfl_xor(mx, 2)); mx = fmaxf(mx, __shfl_xor(mx, 1));
      float e0 = __expf((a0 - mx) * scale), e1 = __expf((a1 - mx) * scale);
      float e2 = __expf((a2 - mx) * scale), e3 = __expf((a3 - mx) * scale);
      float sum = e0 + e1 + e2 + e3;
      sum += __shfl_xor(sum, 8); sum += __shfl_xor(sum, 4);
      sum += __shfl_xor(sum, 2); sum += __shfl_xor(sum, 1);
      float inv = 1.f / sum;
      int row = j + ((lane >> 4) << 3), col = lane & 15;
      pbuf[hd][row][ 0 + col] = (half_t)(e0 * inv);
      pbuf[hd][row][16 + col] = (half_t)(e1 * inv);
      pbuf[hd][row][32 + col] = (half_t)(e2 * inv);
      pbuf[hd][row][48 + col] = (half_t)(e3 * inv);
    }

    // O(16x32) = P(16x64) x Vh(64x32)
    v8f o0 = {}, o1 = {};
    #pragma unroll
    for (int kc = 0; kc < 2; kc++) {
      v16h pa = load_afrag(&pbuf[hd][0][0], 64, 0, kc * 32);
      o0 = wmma16(pa, load_bfrag(&vT[hd][0][0], 64,  0, kc * 32), o0);
      o1 = wmma16(pa, load_bfrag(&vT[hd][0][0], 64, 16, kc * 32), o1);
    }
    int col = lane & 15;
    #pragma unroll
    for (int j = 0; j < 8; j++) {
      int row = base + qr * 16 + j + ((lane >> 4) << 3);
      o16[(size_t)row * GD + hd * GHD + col]      = (half_t)o0[j];
      o16[(size_t)row * GD + hd * GHD + 16 + col] = (half_t)o1[j];
    }
  }
}

// ---------------- cross attention: 1 wave per target token (K=9, VALU) ----------------
__global__ __launch_bounds__(256) void cross_attn_kernel(const half_t* __restrict__ q16,
                                                         const half_t* __restrict__ k16,
                                                         const half_t* __restrict__ v16,
                                                         half_t* __restrict__ out16) {
  int wave = threadIdx.x >> 5, lane = threadIdx.x & 31;
  int t = blockIdx.x * 8 + wave;
  const float scale = 0.17677669529663687f;
  #pragma unroll
  for (int hd = 0; hd < GHEADS; hd++) {
    float qv = (float)q16[(size_t)t * GD + hd * GHD + lane];
    float s[GK];
    #pragma unroll
    for (int j = 0; j < GK; j++) {
      float kv = (float)k16[(size_t)(t * GK + j) * GD + hd * GHD + lane];
      s[j] = wave_sum32(qv * kv) * scale;
    }
    float mx = s[0];
    #pragma unroll
    for (int j = 1; j < GK; j++) mx = fmaxf(mx, s[j]);
    float sum = 0.f;
    #pragma unroll
    for (int j = 0; j < GK; j++) { s[j] = __expf(s[j] - mx); sum += s[j]; }
    float inv = 1.f / sum, o = 0.f;
    #pragma unroll
    for (int j = 0; j < GK; j++)
      o += s[j] * (float)v16[(size_t)(t * GK + j) * GD + hd * GHD + lane];
    out16[(size_t)t * GD + hd * GHD + lane] = (half_t)(o * inv);
  }
}

// ---------------- host orchestration ----------------
extern "C" void kernel_launch(void* const* d_in, const int* in_sizes, int n_in,
                              void* d_out, int out_size, void* d_ws, size_t ws_size,
                              hipStream_t stream) {
  (void)in_sizes; (void)n_in; (void)out_size; (void)ws_size;
  const float* zc_in = (const float*)d_in[1];
  const float* xt    = (const float*)d_in[2];
  const float* zt_in = (const float*)d_in[3];

  struct LinP { const float* b; const float* w; };
  struct LNP  { const float* b; const float* g; };
  struct BlockP { LinP ff1, ff2, k; LNP ln1, ln2; LinP o, q, v; };

  int pi = 4;  // params leaves start; pytree dict keys flatten alphabetically
  auto nxt = [&]() { return (const float*)d_in[pi++]; };
  auto parseBlock = [&]() {
    BlockP bl;
    bl.ff1.b = nxt(); bl.ff1.w = nxt();
    bl.ff2.b = nxt(); bl.ff2.w = nxt();
    bl.k.b   = nxt(); bl.k.w   = nxt();
    bl.ln1.b = nxt(); bl.ln1.g = nxt();
    bl.ln2.b = nxt(); bl.ln2.g = nxt();
    bl.o.b   = nxt(); bl.o.w   = nxt();
    bl.q.b   = nxt(); bl.q.w   = nxt();
    bl.v.b   = nxt(); bl.v.w   = nxt();
    return bl;
  };
  BlockP blk[NLAYERS][2];  // [layer][0]=mhca, [1]=swin (alphabetical)
  for (int L = 0; L < NLAYERS; L++) { blk[L][0] = parseBlock(); blk[L][1] = parseBlock(); }

  // ---- workspace carve ----
  char* wsb = (char*)d_ws; size_t off = 0;
  auto carve = [&](size_t bytes) -> void* {
    void* p = wsb + off; off += (bytes + 255) & ~(size_t)255; return p;
  };
  float* zc   = (float*)carve((size_t)MB * GRID_TOK * GD * 4);
  float* zt   = (float*)carve((size_t)MB * GNT * GD * 4);
  int*   idxb = (int*)  carve((size_t)MB * GNT * GK * 4);
  struct WB { half_t *q, *k, *v, *o, *f1, *f2; };
  WB wb[NLAYERS][2];
  for (int L = 0; L < NLAYERS; L++)
    for (int bI = 0; bI < 2; bI++) {
      wb[L][bI].q  = (half_t*)carve((size_t)GD * GD * 2);
      wb[L][bI].k  = (half_t*)carve((size_t)GD * GD * 2);
      wb[L][bI].v  = (half_t*)carve((size_t)GD * GD * 2);
      wb[L][bI].o  = (half_t*)carve((size_t)GD * GD * 2);
      wb[L][bI].f1 = (half_t*)carve((size_t)GD * GFF * 2);
      wb[L][bI].f2 = (half_t*)carve((size_t)GFF * GD * 2);
    }
  float*  zw  = (float*) carve((size_t)GRID_TOK * GD * 4);
  half_t* a16 = (half_t*)carve((size_t)GRID_TOK * GD * 2);
  half_t* q16 = (half_t*)carve((size_t)GRID_TOK * GD * 2);
  half_t* k16 = (half_t*)carve((size_t)GNT * GK * GD * 2);
  half_t* v16 = (half_t*)carve((size_t)GNT * GK * GD * 2);
  half_t* g16 = (half_t*)carve((size_t)GNT * GK * GD * 2);
  half_t* h16 = (half_t*)carve((size_t)GRID_TOK * GFF * 2);

  hipMemcpyAsync(zc, zc_in, (size_t)MB * GRID_TOK * GD * 4, hipMemcpyDeviceToDevice, stream);
  hipMemcpyAsync(zt, zt_in, (size_t)MB * GNT * GD * 4, hipMemcpyDeviceToDevice, stream);

  // weights -> transposed f16, once
  for (int L = 0; L < NLAYERS; L++)
    for (int bI = 0; bI < 2; bI++) {
      wcast_t<<<(GD * GD + 255) / 256, 256, 0, stream>>>(blk[L][bI].q.w, wb[L][bI].q, GD, GD);
      wcast_t<<<(GD * GD + 255) / 256, 256, 0, stream>>>(blk[L][bI].k.w, wb[L][bI].k, GD, GD);
      wcast_t<<<(GD * GD + 255) / 256, 256, 0, stream>>>(blk[L][bI].v.w, wb[L][bI].v, GD, GD);
      wcast_t<<<(GD * GD + 255) / 256, 256, 0, stream>>>(blk[L][bI].o.w, wb[L][bI].o, GD, GD);
      wcast_t<<<(GD * GFF + 255) / 256, 256, 0, stream>>>(blk[L][bI].ff1.w, wb[L][bI].f1, GD, GFF);
      wcast_t<<<(GFF * GD + 255) / 256, 256, 0, stream>>>(blk[L][bI].ff2.w, wb[L][bI].f2, GFF, GD);
    }
  idx_kernel<<<(MB * GNT + 255) / 256, 256, 0, stream>>>(xt, idxb);

  for (int m = 0; m < MB; m++) {
    float* zc_m = zc + (size_t)m * GRID_TOK * GD;
    float* zt_m = zt + (size_t)m * GNT * GD;
    const int* idx_m = idxb + (size_t)m * GNT * GK;
    for (int L = 0; L < NLAYERS; L++) {
      const BlockP& sw = blk[L][1]; const WB& sww = wb[L][1];
      const BlockP& mc = blk[L][0]; const WB& mcw = wb[L][0];

      // -------- SWIN window attention, shift 0 then WS/2 --------
      for (int pass = 0; pass < 2; pass++) {
        int shift = pass ? (GWS / 2) : 0;
        dim3 gDD(GD / 64, GRID_TOK / 32);
        win_gather_ln<<<GRID_TOK / 8, 256, 0, stream>>>(zc_m, zw, a16, sw.ln1.g, sw.ln1.b, shift);
        gemm_wmma<MODE_F16, GD><<<gDD, 32, 0, stream>>>(a16, sww.q, sw.q.b, GD, q16, nullptr);
        gemm_wmma<MODE_F16, GD><<<gDD, 32, 0, stream>>>(a16, sww.k, sw.k.b, GD, k16, nullptr);
        gemm_wmma<MODE_F16, GD><<<gDD, 32, 0, stream>>>(a16, sww.v, sw.v.b, GD, v16, nullptr);
        window_attn_kernel<<<GRID_TOK / 64, 256, 0, stream>>>(q16, k16, v16, a16);
        gemm_wmma<MODE_RES, GD><<<gDD, 32, 0, stream>>>(a16, sww.o, sw.o.b, GD, nullptr, zw);
        ln_rows<<<GRID_TOK / 8, 256, 0, stream>>>(zw, sw.ln2.g, sw.ln2.b, a16);
        gemm_wmma<MODE_GELU, GD><<<dim3(GFF / 64, GRID_TOK / 32), 32, 0, stream>>>(
            a16, sww.f1, sw.ff1.b, GFF, h16, nullptr);
        gemm_wmma<MODE_RES, GFF><<<dim3(GD / 64, GRID_TOK / 32), 32, 0, stream>>>(
            h16, sww.f2, sw.ff2.b, GD, nullptr, zw);
        win_scatter<<<GRID_TOK, 256, 0, stream>>>(zw, zc_m, shift);
      }

      // -------- top-K gather cross attention on targets --------
      gather_ln_kv<<<(GNT * GK) / 8, 256, 0, stream>>>(zc_m, idx_m, mc.ln1.g, mc.ln1.b, g16);
      ln_rows<<<GNT / 8, 256, 0, stream>>>(zt_m, mc.ln1.g, mc.ln1.b, a16);
      gemm_wmma<MODE_F16, GD><<<dim3(GD / 64, GNT / 32), 32, 0, stream>>>(
          a16, mcw.q, mc.q.b, GD, q16, nullptr);
      gemm_wmma<MODE_F16, GD><<<dim3(GD / 64, (GNT * GK) / 32), 32, 0, stream>>>(
          g16, mcw.k, mc.k.b, GD, k16, nullptr);
      gemm_wmma<MODE_F16, GD><<<dim3(GD / 64, (GNT * GK) / 32), 32, 0, stream>>>(
          g16, mcw.v, mc.v.b, GD, v16, nullptr);
      cross_attn_kernel<<<GNT / 8, 256, 0, stream>>>(q16, k16, v16, a16);
      gemm_wmma<MODE_RES, GD><<<dim3(GD / 64, GNT / 32), 32, 0, stream>>>(
          a16, mcw.o, mc.o.b, GD, nullptr, zt_m);
      ln_rows<<<GNT / 8, 256, 0, stream>>>(zt_m, mc.ln2.g, mc.ln2.b, a16);
      gemm_wmma<MODE_GELU, GD><<<dim3(GFF / 64, GNT / 32), 32, 0, stream>>>(
          a16, mcw.f1, mc.ff1.b, GFF, h16, nullptr);
      gemm_wmma<MODE_RES, GFF><<<dim3(GD / 64, GNT / 32), 32, 0, stream>>>(
          h16, mcw.f2, mc.ff2.b, GD, nullptr, zt_m);
    }
  }

  hipMemcpyAsync(d_out, zt, (size_t)MB * GNT * GD * 4, hipMemcpyDeviceToDevice, stream);
}
